// TrafficGATv2_40681930227901
// MI455X (gfx1250) — compile-verified, matrix operand order checked by reading, not run on hardware
//
#include <hip/hip_runtime.h>
#include <hip/hip_bf16.h>

#define NN   20000      // nodes
#define NE   320000     // edges
#define NESL 340000     // edges + self loops
#define HID  64
#define HEADS 4
#define NEG_SLOPE 0.2f

typedef __attribute__((ext_vector_type(16))) _Float16 v16h;
typedef __attribute__((ext_vector_type(8)))  float    v8f;

// ---------- helpers ----------
__device__ __forceinline__ unsigned f2ord(float f) {
    unsigned u = __float_as_uint(f);
    return (u & 0x80000000u) ? ~u : (u | 0x80000000u);
}
__device__ __forceinline__ float ord2f(unsigned u) {
    return __uint_as_float((u & 0x80000000u) ? (u & 0x7FFFFFFFu) : ~u);
}

// ---------- fills ----------
__global__ void k_fill_f(float* p, float v, int n) {
    int i = blockIdx.x * blockDim.x + threadIdx.x;
    if (i < n) p[i] = v;
}
__global__ void k_fill_u(unsigned* p, unsigned v, int n) {
    int i = blockIdx.x * blockDim.x + threadIdx.x;
    if (i < n) p[i] = v;
}

// ---------- input embeddings (tiny K: VALU) ----------
__global__ void k_embed_node(const float* __restrict__ x, const float* __restrict__ W,
                             const float* __restrict__ b, float* __restrict__ out) {
    int idx = blockIdx.x * blockDim.x + threadIdx.x;
    if (idx >= NN * HID) return;
    int n = idx >> 6, c = idx & 63;
    float s = b[c];
#pragma unroll
    for (int k = 0; k < 4; ++k) s += x[n * 4 + k] * W[k * HID + c];
    out[idx] = s;
}
__global__ void k_embed_edge(const float* __restrict__ ef, const float* __restrict__ W,
                             const float* __restrict__ b, float* __restrict__ out) {
    int idx = blockIdx.x * blockDim.x + threadIdx.x;
    if (idx >= NE * HID) return;
    int e = idx >> 6, c = idx & 63;
    float s = b[c];
#pragma unroll
    for (int k = 0; k < 3; ++k) s += ef[e * 3 + k] * W[k * HID + c];
    out[idx] = s;
}

// ---------- WMMA GEMM: out[M,256] = h[M,64] @ W[64,256] + b ----------
// grid: (M/16, 2), block: 256 (8 waves); each wave one 16x16 tile, K=64 in 2 slices.
__global__ void k_gemm_hw(const float* __restrict__ h, const float* __restrict__ W,
                          const float* __restrict__ b, float* __restrict__ out) {
    __shared__ _Float16 aT[16][64];
    int tid = threadIdx.x;
    int m0 = blockIdx.x * 16;
    for (int i = tid; i < 16 * 64; i += 256) {
        int r = i >> 6, c = i & 63;
        aT[r][c] = (_Float16)h[(m0 + r) * 64 + c];
    }
    __syncthreads();
    int wave = tid >> 5, lane = tid & 31;
    int n0 = (blockIdx.y * 8 + wave) * 16;
    int mrow = lane & 15, kg = lane >> 4;          // kg: K-group select (wave32 halves)
    v8f acc = {};
#pragma unroll
    for (int ks = 0; ks < 2; ++ks) {
        int k0 = ks * 32;
        v16h a, bf;
#pragma unroll
        for (int v = 0; v < 8; ++v) {
            int kb = k0 + ((v & 4) ? 16 : 0) + kg * 8 + ((v & 3) << 1);
            a[2 * v]     = aT[mrow][kb];
            a[2 * v + 1] = aT[mrow][kb + 1];
            bf[2 * v]     = (_Float16)W[kb * 256 + n0 + mrow];
            bf[2 * v + 1] = (_Float16)W[(kb + 1) * 256 + n0 + mrow];
        }
        acc = __builtin_amdgcn_wmma_f32_16x16x32_f16(false, a, false, bf, (short)0, acc,
                                                     false, false);
    }
    int ncol = lane & 15, rbase = (lane >> 4) * 8;
    float bc = b[n0 + ncol];
#pragma unroll
    for (int r = 0; r < 8; ++r)
        out[(m0 + rbase + r) * 256 + n0 + ncol] = acc[r] + bc;
}

// ---------- attention logits + segment max ----------
__global__ void k_logits(const float* __restrict__ xl, const float* __restrict__ xr,
                         const int* __restrict__ src, const int* __restrict__ tgt,
                         const float* __restrict__ att, float* __restrict__ logits,
                         unsigned* __restrict__ maxOrd) {
    int idx = blockIdx.x * blockDim.x + threadIdx.x;
    if (idx >= NESL * HEADS) return;
    int e = idx >> 2, h = idx & 3;
    int s = (e < NE) ? src[e] : (e - NE);
    int t = (e < NE) ? tgt[e] : (e - NE);
    const float* pl = xl + s * 256 + h * 64;
    const float* pr = xr + t * 256 + h * 64;
    const float* pa = att + h * 64;
    float acc = 0.f;
#pragma unroll 8
    for (int c = 0; c < 64; ++c) {
        float v = pl[c] + pr[c];
        v = v > 0.f ? v : NEG_SLOPE * v;
        acc += v * pa[c];
    }
    logits[idx] = acc;
    atomicMax(&maxOrd[t * HEADS + h], f2ord(acc));
}

// ---------- exp + segment sum ----------
__global__ void k_expsum(float* __restrict__ logits, const unsigned* __restrict__ maxOrd,
                         float* __restrict__ denom, const int* __restrict__ tgt) {
    int idx = blockIdx.x * blockDim.x + threadIdx.x;
    if (idx >= NESL * HEADS) return;
    int e = idx >> 2, h = idx & 3;
    int t = (e < NE) ? tgt[e] : (e - NE);
    float ex = __expf(logits[idx] - ord2f(maxOrd[t * HEADS + h]));
    logits[idx] = ex;
    atomicAdd(&denom[t * HEADS + h], ex);
}

// ---------- weighted scatter ----------
__global__ void k_scatter(const float* __restrict__ xl, const float* __restrict__ ex,
                          const float* __restrict__ denom, const int* __restrict__ src,
                          const int* __restrict__ tgt, float* __restrict__ accum) {
    long idx = (long)blockIdx.x * blockDim.x + threadIdx.x;
    if (idx >= (long)NESL * HEADS * HID) return;
    int c = (int)(idx & 63);
    int h = (int)((idx >> 6) & 3);
    int e = (int)(idx >> 8);
    int s = (e < NE) ? src[e] : (e - NE);
    int t = (e < NE) ? tgt[e] : (e - NE);
    float alpha = ex[e * HEADS + h] / denom[t * HEADS + h];
    atomicAdd(&accum[t * 256 + h * 64 + c], xl[s * 256 + h * 64 + c] * alpha);
}

// ---------- head mean + bias (+ELU) ----------
__global__ void k_headmean(const float* __restrict__ accum, const float* __restrict__ bias,
                           float* __restrict__ node_emb, int do_elu) {
    int idx = blockIdx.x * blockDim.x + threadIdx.x;
    if (idx >= NN * HID) return;
    int n = idx >> 6, c = idx & 63;
    float s = 0.f;
#pragma unroll
    for (int h = 0; h < HEADS; ++h) s += accum[n * 256 + h * 64 + c];
    s = s * 0.25f + bias[c];
    if (do_elu) s = s > 0.f ? s : (__expf(s) - 1.f);
    node_emb[idx] = s;
}

// ---------- final edge MLP with WMMA: 192->64->32->1 ----------
// block = 128 threads (4 waves); each wave processes 16 edges.
__global__ void k_mlp(const float* __restrict__ node_emb, const float* __restrict__ edge_emb,
                      const int* __restrict__ src, const int* __restrict__ tgt,
                      const float* __restrict__ W1, const float* __restrict__ b1,
                      const float* __restrict__ W2, const float* __restrict__ b2,
                      const float* __restrict__ W3, const float* __restrict__ b3,
                      float* __restrict__ out) {
    __shared__ _Float16 ctx[4][16][192];
    __shared__ _Float16 h1s[4][16][64];
    __shared__ float    h2s[4][16][32];
    int tid = threadIdx.x, wave = tid >> 5, lane = tid & 31;
    int e0 = (blockIdx.x * 4 + wave) * 16;

    for (int i = lane; i < 16 * 192; i += 32) {
        int el = i / 192, c = i - el * 192;
        int e = e0 + el;
        float v;
        if (c < 64)       v = node_emb[src[e] * HID + c];
        else if (c < 128) v = node_emb[tgt[e] * HID + (c - 64)];
        else              v = edge_emb[e * HID + (c - 128)];
        ctx[wave][el][c] = (_Float16)v;
    }
    __syncthreads();

    int mrow = lane & 15, kg = lane >> 4;
    int ncol = lane & 15, rbase = (lane >> 4) * 8;

    // layer 1: [16,192] @ [192,64]
#pragma unroll
    for (int nt = 0; nt < 4; ++nt) {
        v8f acc = {};
#pragma unroll
        for (int ks = 0; ks < 6; ++ks) {
            int k0 = ks * 32;
            v16h a, bf;
#pragma unroll
            for (int v = 0; v < 8; ++v) {
                int kb = k0 + ((v & 4) ? 16 : 0) + kg * 8 + ((v & 3) << 1);
                a[2 * v]     = ctx[wave][mrow][kb];
                a[2 * v + 1] = ctx[wave][mrow][kb + 1];
                bf[2 * v]     = (_Float16)W1[kb * 64 + nt * 16 + mrow];
                bf[2 * v + 1] = (_Float16)W1[(kb + 1) * 64 + nt * 16 + mrow];
            }
            acc = __builtin_amdgcn_wmma_f32_16x16x32_f16(false, a, false, bf, (short)0, acc,
                                                         false, false);
        }
#pragma unroll
        for (int r = 0; r < 8; ++r) {
            float v = acc[r] + b1[nt * 16 + ncol];
            h1s[wave][rbase + r][nt * 16 + ncol] = (_Float16)fmaxf(v, 0.f);
        }
    }
    __syncthreads();

    // layer 2: [16,64] @ [64,32]
#pragma unroll
    for (int nt = 0; nt < 2; ++nt) {
        v8f acc = {};
#pragma unroll
        for (int ks = 0; ks < 2; ++ks) {
            int k0 = ks * 32;
            v16h a, bf;
#pragma unroll
            for (int v = 0; v < 8; ++v) {
                int kb = k0 + ((v & 4) ? 16 : 0) + kg * 8 + ((v & 3) << 1);
                a[2 * v]     = h1s[wave][mrow][kb];
                a[2 * v + 1] = h1s[wave][mrow][kb + 1];
                bf[2 * v]     = (_Float16)W2[kb * 32 + nt * 16 + mrow];
                bf[2 * v + 1] = (_Float16)W2[(kb + 1) * 32 + nt * 16 + mrow];
            }
            acc = __builtin_amdgcn_wmma_f32_16x16x32_f16(false, a, false, bf, (short)0, acc,
                                                         false, false);
        }
#pragma unroll
        for (int r = 0; r < 8; ++r)
            h2s[wave][rbase + r][nt * 16 + ncol] = fmaxf(acc[r] + b2[nt * 16 + ncol], 0.f);
    }
    __syncthreads();

    // layer 3: [16,32] @ [32,1] on lanes 0-15
    if (kg == 0) {
        float s = b3[0];
#pragma unroll
        for (int c = 0; c < 32; ++c) s += h2s[wave][mrow][c] * W3[c];
        out[e0 + mrow] = fmaxf(s, 0.f);
    }
}

// ---------- host launcher ----------
extern "C" void kernel_launch(void* const* d_in, const int* in_sizes, int n_in,
                              void* d_out, int out_size, void* d_ws, size_t ws_size,
                              hipStream_t stream) {
    // setup_inputs() flattening (dict insertion order, depth-first):
    const float* x       = (const float*)d_in[0];   // [20000,4]
    const int*   ei      = (const int*)d_in[1];     // [2,320000]
    const float* ef      = (const float*)d_in[2];   // [320000,3]
    const float* node_W  = (const float*)d_in[3];
    const float* node_b  = (const float*)d_in[4];
    const float* edge_W  = (const float*)d_in[5];
    const float* edge_b  = (const float*)d_in[6];
    // gat layers: 7 + l*6 : {Wl, bl, Wr, br, att, bias}
    const float* mlp_W1  = (const float*)d_in[25];
    const float* mlp_b1  = (const float*)d_in[26];
    const float* mlp_W2  = (const float*)d_in[27];
    const float* mlp_b2  = (const float*)d_in[28];
    const float* mlp_W3  = (const float*)d_in[29];
    const float* mlp_b3  = (const float*)d_in[30];

    const int* src = ei;
    const int* tgt = ei + NE;

    float* ws = (float*)d_ws;
    float* node_emb = ws;  ws += NN * HID;
    float* edge_emb = ws;  ws += NE * HID;
    float* xl       = ws;  ws += NN * 256;
    float* xr       = ws;  ws += NN * 256;
    float* logits   = ws;  ws += NESL * HEADS;
    float* denom    = ws;  ws += NN * HEADS;
    unsigned* maxOrd = (unsigned*)ws; ws += NN * HEADS;
    float* accum    = ws;  ws += NN * 256;

    dim3 b256(256);
    k_embed_node<<<(NN * HID + 255) / 256, b256, 0, stream>>>(x, node_W, node_b, node_emb);
    k_embed_edge<<<(NE * HID + 255) / 256, b256, 0, stream>>>(ef, edge_W, edge_b, edge_emb);

    for (int l = 0; l < 3; ++l) {
        const float* Wl   = (const float*)d_in[7 + l * 6 + 0];
        const float* bl   = (const float*)d_in[7 + l * 6 + 1];
        const float* Wr   = (const float*)d_in[7 + l * 6 + 2];
        const float* br   = (const float*)d_in[7 + l * 6 + 3];
        const float* att  = (const float*)d_in[7 + l * 6 + 4];
        const float* bias = (const float*)d_in[7 + l * 6 + 5];

        k_gemm_hw<<<dim3(NN / 16, 2), b256, 0, stream>>>(node_emb, Wl, bl, xl);
        k_gemm_hw<<<dim3(NN / 16, 2), b256, 0, stream>>>(node_emb, Wr, br, xr);

        k_fill_f<<<(NN * HEADS + 255) / 256, b256, 0, stream>>>(denom, 0.f, NN * HEADS);
        k_fill_u<<<(NN * HEADS + 255) / 256, b256, 0, stream>>>(maxOrd, 0u, NN * HEADS);
        k_fill_f<<<(NN * 256 + 255) / 256, b256, 0, stream>>>(accum, 0.f, NN * 256);

        k_logits<<<(NESL * HEADS + 255) / 256, b256, 0, stream>>>(xl, xr, src, tgt, att,
                                                                  logits, maxOrd);
        k_expsum<<<(NESL * HEADS + 255) / 256, b256, 0, stream>>>(logits, maxOrd, denom, tgt);

        long scatter_n = (long)NESL * HEADS * HID;
        k_scatter<<<(int)((scatter_n + 255) / 256), b256, 0, stream>>>(xl, logits, denom,
                                                                       src, tgt, accum);
        k_headmean<<<(NN * HID + 255) / 256, b256, 0, stream>>>(accum, bias, node_emb,
                                                                l < 2 ? 1 : 0);
    }

    k_mlp<<<NE / 64, 128, 0, stream>>>(node_emb, edge_emb, src, tgt,
                                       mlp_W1, mlp_b1, mlp_W2, mlp_b2, mlp_W3, mlp_b3,
                                       (float*)d_out);
}
